// LSTMModel_78348793413868
// MI455X (gfx1250) — matrix operand order repaired and checked by reading, not compiled, not run on hardware
//
#include <hip/hip_runtime.h>

// ----------------------------------------------------------------------------
// Fused 2-layer LSTM + FC head for MI455X (gfx1250, wave32, WMMA).
//
//  * grid = 32 WGs x 256 threads (8 waves); each WG owns 16 batch rows and
//    runs the full T=512 recurrence for both layers (phases) -> no global sync.
//  * All weights LDS-resident in f16, pre-swizzled to WMMA B-fragment order
//    (phase A 192 KB, phase B 256 KB; LDS total 282 KB < 320 KB).
//  * Wave w owns gate columns [16w,16w+16) of i/f/g/o -> register-only cell
//    update; c-state is a persistent v8f per lane; h written back to LDS in
//    A-fragment order (4 KB) for the next step's h @ W_hh.T.
//  * Double-buffered input staging: phase A overlaps fp32 x loads with WMMA
//    via register staging; phase B DMAs the f16 h1 stream straight into
//    fragment layout with global_load_async_to_lds_b32 (ASYNCcnt).
//  * Native v_tanh_f32 / v_exp_f32 / v_rcp_f32 for the gate nonlinearities.
// ----------------------------------------------------------------------------

#define HH   128     // hidden size
#define GG   512     // 4*H gate width
#define TT   512     // sequence length
#define BB   512     // batch
#define DD   37      // input features / classes
#define NCLS 37
#define BT   16      // batch rows per workgroup
#define NTHR 256

typedef __attribute__((ext_vector_type(16))) _Float16 v16h;
typedef __attribute__((ext_vector_type(8)))  float    v8f;

// LDS layout (byte offsets)
#define OFF_WIN   0                     // input->gate weights (<=128 KB)
#define OFF_WHH   (128*1024)            // hidden->gate weights (128 KB)
#define OFF_HA    (256*1024)            // h in A-fragment order (4 KB)
#define OFF_XA    (OFF_HA + 4096)       // input tile, 2 parity buffers (8 KB)
#define OFF_BIAS  (OFF_XA + 8192)       // b_ih+b_hh (2 KB)
#define OFF_HF    (OFF_BIAS + 2048)     // final h2, fp32 row-major (8 KB)
#define OFF_Y1    (OFF_HF + 8192)       // fc1 activations (4 KB)
#define SMEM_BYTES (OFF_Y1 + 4096)      // 288768 B < 320 KB

__device__ __forceinline__ float fast_tanh(float x) {
#if __has_builtin(__builtin_amdgcn_tanhf)
    return __builtin_amdgcn_tanhf(x);          // v_tanh_f32
#else
    return tanhf(x);
#endif
}
__device__ __forceinline__ float sigmoidf_(float v) {
    // 1/(1+e^-v) = rcp(1 + 2^(-v*log2(e)))  -> v_exp_f32 + v_rcp_f32
    return __builtin_amdgcn_rcpf(1.0f + __builtin_amdgcn_exp2f(-1.44269504088896f * v));
}

// Position of element (m, k) inside one 16x32 f16 A-fragment chunk
// (ISA 7.12.2: lanes 0-15 -> M, K 0..7 & 16..23; lanes 16-31 -> K 8..15 & 24..31)
__device__ __forceinline__ int afrag_pos(int m, int k32) {
    int half = (k32 >> 3) & 1;
    int lane = m + (half << 4);
    int e    = (k32 & 7) + ((k32 >> 4) << 3);
    return lane * 16 + e;
}

// Convert W[G][Ksrc] (row-major fp32, gate-major) into f16 B-fragment-linear
// LDS layout: flat = ((nt*KC + kc)*32 + lane)*16 + e; within a 32x16 B tile:
// n = lane%16, k = (e&7) + 8*(lane/16) + 16*(e/8).  K padded to KC*32 with zeros.
template<int KC>
__device__ void load_wfrag(_Float16* dst, const float* __restrict__ W, int Ksrc) {
    constexpr int LOGKC = (KC == 2) ? 1 : 2;
    const int total = 32 * KC * 512;
    for (int idx = (int)threadIdx.x; idx < total; idx += NTHR) {
        int e    = idx & 15;
        int lane = (idx >> 4) & 31;
        int kcnt = idx >> 9;
        int kc   = kcnt & (KC - 1);
        int nt   = kcnt >> LOGKC;
        int n = nt * 16 + (lane & 15);
        int k = kc * 32 + (e & 7) + ((lane >> 4) << 3) + ((e >> 3) << 4);
        float v = (k < Ksrc) ? W[n * Ksrc + k] : 0.0f;
        dst[idx] = (_Float16)v;
    }
}

__device__ void zero_hA(char* smem) {
    _Float16* hA = (_Float16*)(smem + OFF_HA);
    for (int idx = (int)threadIdx.x; idx < 4 * 512; idx += NTHR)
        hA[idx] = (_Float16)0.0f;
}

// One LSTM layer over all T steps for this WG's 16 batch rows.
template<int KCIN, bool FROM_X, bool WRITE_H1, bool SAVE_HF>
__device__ void run_layer(char* smem,
                          const float*    __restrict__ xin,
                          const _Float16* __restrict__ hin_ws,
                          _Float16*       __restrict__ hout_ws,
                          int bBase)
{
    _Float16* wIn  = (_Float16*)(smem + OFF_WIN);
    _Float16* wHh  = (_Float16*)(smem + OFF_WHH);
    _Float16* hA   = (_Float16*)(smem + OFF_HA);
    _Float16* xA   = (_Float16*)(smem + OFF_XA);   // 2 buffers of 2048 elems
    float*    bias = (float*)(smem + OFF_BIAS);
    float*    hF   = (float*)(smem + OFF_HF);

    const int tid   = (int)threadIdx.x;
    const int lane  = tid & 31;
    const int w     = tid >> 5;          // wave id 0..7
    const int colq  = lane & 15;         // C-layout column within tile
    const int rhalf = lane >> 4;         // C-layout row-half
    const int col   = w * 16 + colq;     // owned h column

    // ---- prestage t=0 into xA buffer 0 (+ zero padding in both buffers) ----
    if (FROM_X) {
        for (int idx = tid; idx < BT * 64; idx += NTHR) {
            int m = idx >> 6, k = idx & 63;
            int pos = (k >> 5) * 512 + afrag_pos(m, k & 31);
            float v = (k < DD) ? xin[((size_t)(bBase + m) * TT) * DD + k] : 0.0f;
            xA[pos] = (_Float16)v;
            if (k >= DD) xA[2048 + pos] = (_Float16)0.0f;   // pad buffer 1 once
        }
    } else {
        for (int idx = tid; idx < BT * 128; idx += NTHR) {
            int m = idx >> 7, k = idx & 127;
            xA[(k >> 5) * 512 + afrag_pos(m, k & 31)] =
                hin_ws[((size_t)(bBase + m) * TT) * HH + k];
        }
    }

    // ---- per-lane address state for async staging (phase B) ----
    unsigned voff[4] = {0, 0, 0, 0};
    unsigned ldsd[4] = {0, 0, 0, 0};
    if (!FROM_X) {
        #pragma unroll
        for (int j = 0; j < 4; ++j) {
            int q = w * 128 + j * 32 + lane;       // pair index 0..1023
            int m = q >> 6;
            int k = (q & 63) * 2;                  // even k -> contiguous pair
            voff[j] = (unsigned)((((size_t)(bBase + m) * TT + 1) * HH + k) * 2u);
            ldsd[j] = (unsigned)(OFF_XA + ((k >> 5) * 512 + afrag_pos(m, k & 31)) * 2);
        }
    }
    __syncthreads();

    v8f cstate;
    #pragma unroll
    for (int r = 0; r < 8; ++r) cstate[r] = 0.0f;

    for (int t = 0; t < TT; ++t) {
        const int p = t & 1;
        _Float16* xAp = xA + p * 2048;

        // ---- A fragments (shared across this wave's 4 N-tiles) ----
        v16h ah[4];
        #pragma unroll
        for (int kc = 0; kc < 4; ++kc)
            ah[kc] = *(const v16h*)(hA + kc * 512 + lane * 16);
        v16h ax[KCIN];
        #pragma unroll
        for (int kc = 0; kc < KCIN; ++kc)
            ax[kc] = *(const v16h*)(xAp + kc * 512 + lane * 16);

        // ---- kick off staging of step t+1 (overlaps with WMMA below) ----
        float sv[4] = {0.0f, 0.0f, 0.0f, 0.0f};
        if (FROM_X) {
            if (t + 1 < TT) {
                #pragma unroll
                for (int i = 0; i < 4; ++i) {
                    int idx = tid + i * NTHR;
                    int m = idx >> 6, k = idx & 63;
                    if (k < DD)
                        sv[i] = xin[((size_t)(bBase + m) * TT + (t + 1)) * DD + k];
                }
            }
        } else {
            if (t + 1 < TT) {
                const unsigned lpar = (unsigned)((p ^ 1) * 4096);
                #pragma unroll
                for (int j = 0; j < 4; ++j) {
                    unsigned ld = ldsd[j] + lpar;
                    asm volatile("global_load_async_to_lds_b32 %0, %1, %2"
                                 :: "v"(ld), "v"(voff[j]),
                                    "s"((unsigned long long)(uintptr_t)hin_ws)
                                 : "memory");
                    voff[j] += (unsigned)(HH * 2);
                }
            }
        }
        __syncthreads();   // MID: all hA/xA[p] fragment reads complete

        // ---- gate GEMMs: wave w computes N-tiles {w, w+8, w+16, w+24} ----
        v8f gate[4];
        #pragma unroll
        for (int g = 0; g < 4; ++g) {
            const int nt = 8 * g + w;
            const float bv = bias[nt * 16 + colq];
            v8f acc;
            #pragma unroll
            for (int r = 0; r < 8; ++r) acc[r] = bv;
            #pragma unroll
            for (int kc = 0; kc < KCIN; ++kc) {
                v16h b = *(const v16h*)(wIn + ((nt * KCIN + kc) * 32 + lane) * 16);
                acc = __builtin_amdgcn_wmma_f32_16x16x32_f16(
                        false, ax[kc], false, b, (short)0, acc, false, false);
            }
            #pragma unroll
            for (int kc = 0; kc < 4; ++kc) {
                v16h b = *(const v16h*)(wHh + ((nt * 4 + kc) * 32 + lane) * 16);
                acc = __builtin_amdgcn_wmma_f32_16x16x32_f16(
                        false, ah[kc], false, b, (short)0, acc, false, false);
            }
            gate[g] = acc;
        }

        // ---- commit register-staged x_{t+1} (phase A) ----
        if (FROM_X && t + 1 < TT) {
            _Float16* xAn = xA + (p ^ 1) * 2048;
            #pragma unroll
            for (int i = 0; i < 4; ++i) {
                int idx = tid + i * NTHR;
                int m = idx >> 6, k = idx & 63;
                if (k < DD)
                    xAn[(k >> 5) * 512 + afrag_pos(m, k & 31)] = (_Float16)sv[i];
            }
        }

        // ---- register-only LSTM cell update for owned columns ----
        #pragma unroll
        for (int r = 0; r < 8; ++r) {
            float iv = sigmoidf_(gate[0][r]);
            float fv = sigmoidf_(gate[1][r]);
            float gv = fast_tanh(gate[2][r]);
            float ov = sigmoidf_(gate[3][r]);
            float c  = fv * cstate[r] + iv * gv;
            cstate[r] = c;
            float h = ov * fast_tanh(c);
            int m = r + 8 * rhalf;
            hA[(col >> 5) * 512 + afrag_pos(m, col & 31)] = (_Float16)h;
            if (WRITE_H1)
                hout_ws[((size_t)(bBase + m) * TT + t) * HH + col] = (_Float16)h;
            if (SAVE_HF && t == TT - 1)
                hF[m * HH + col] = h;
        }

        if (!FROM_X)
            asm volatile("s_wait_asynccnt 0x0" ::: "memory");
        __syncthreads();   // END: hA + staged xA visible for step t+1
    }
    __syncthreads();
}

extern "C" __global__ void __launch_bounds__(NTHR)
lstm2_fused(const float* __restrict__ x,
            const float* __restrict__ Wih0, const float* __restrict__ Whh0,
            const float* __restrict__ bih0, const float* __restrict__ bhh0,
            const float* __restrict__ Wih1, const float* __restrict__ Whh1,
            const float* __restrict__ bih1, const float* __restrict__ bhh1,
            const float* __restrict__ fc1w, const float* __restrict__ fc1b,
            const float* __restrict__ fc2w, const float* __restrict__ fc2b,
            float* __restrict__ out, _Float16* __restrict__ h1ws)
{
    extern __shared__ char smem[];
    const int tid   = (int)threadIdx.x;
    const int bBase = (int)blockIdx.x * BT;
    float* bias = (float*)(smem + OFF_BIAS);

    // ---------------- Phase A: layer 0 ----------------
    load_wfrag<2>((_Float16*)(smem + OFF_WIN), Wih0, DD);   // 512x64 (pad 37->64)
    load_wfrag<4>((_Float16*)(smem + OFF_WHH), Whh0, HH);   // 512x128
    for (int idx = tid; idx < GG; idx += NTHR) bias[idx] = bih0[idx] + bhh0[idx];
    zero_hA(smem);
    __syncthreads();
    run_layer<2, true, true, false>(smem, x, nullptr, h1ws, bBase);

    __threadfence();           // make this WG's h1 stores visible to all waves
    __syncthreads();

    // ---------------- Phase B: layer 1 ----------------
    load_wfrag<4>((_Float16*)(smem + OFF_WIN), Wih1, HH);
    load_wfrag<4>((_Float16*)(smem + OFF_WHH), Whh1, HH);
    for (int idx = tid; idx < GG; idx += NTHR) bias[idx] = bih1[idx] + bhh1[idx];
    zero_hA(smem);
    __syncthreads();
    run_layer<4, false, false, true>(smem, nullptr, h1ws, nullptr, bBase);

    // ---------------- FC head on h2[:, -1, :] ----------------
    float* hF = (float*)(smem + OFF_HF);
    float* y1 = (float*)(smem + OFF_Y1);
    for (int idx = tid; idx < BT * 64; idx += NTHR) {      // relu(h @ fc1.T + b)
        int m = idx >> 6, j = idx & 63;
        float s = fc1b[j];
        #pragma unroll 4
        for (int k = 0; k < HH; ++k) s += hF[m * HH + k] * fc1w[j * HH + k];
        y1[idx] = fmaxf(s, 0.0f);
    }
    __syncthreads();
    for (int idx = tid; idx < BT * NCLS; idx += NTHR) {    // y1 @ fc2.T + b
        int m = idx / NCLS, j = idx - m * NCLS;
        float s = fc2b[j];
        #pragma unroll 4
        for (int k = 0; k < 64; ++k) s += y1[m * 64 + k] * fc2w[j * 64 + k];
        out[(size_t)(bBase + m) * NCLS + j] = s;
    }
}

extern "C" void kernel_launch(void* const* d_in, const int* in_sizes, int n_in,
                              void* d_out, int out_size, void* d_ws, size_t ws_size,
                              hipStream_t stream) {
    (void)in_sizes; (void)n_in; (void)out_size; (void)ws_size;
    const float* x    = (const float*)d_in[0];
    const float* Wih0 = (const float*)d_in[1];
    const float* Whh0 = (const float*)d_in[2];
    const float* bih0 = (const float*)d_in[3];
    const float* bhh0 = (const float*)d_in[4];
    const float* Wih1 = (const float*)d_in[5];
    const float* Whh1 = (const float*)d_in[6];
    const float* bih1 = (const float*)d_in[7];
    const float* bhh1 = (const float*)d_in[8];
    const float* fc1w = (const float*)d_in[9];
    const float* fc1b = (const float*)d_in[10];
    const float* fc2w = (const float*)d_in[11];
    const float* fc2b = (const float*)d_in[12];
    float*    out  = (float*)d_out;
    _Float16* h1ws = (_Float16*)d_ws;   // needs B*T*H*2 = 64 MiB scratch

    lstm2_fused<<<dim3(BB / BT), dim3(NTHR), SMEM_BYTES, stream>>>(
        x, Wih0, Whh0, bih0, bhh0, Wih1, Whh1, bih1, bhh1,
        fc1w, fc1b, fc2w, fc2b, out, h1ws);
}
